// SelfAttentionBlock_21388937134616
// MI455X (gfx1250) — matrix operand order repaired and checked by reading, not compiled, not run on hardware
//
#include <hip/hip_runtime.h>
#include <hip/hip_bf16.h>
#include <math.h>

typedef __attribute__((ext_vector_type(16))) _Float16 v16h;
typedef __attribute__((ext_vector_type(8)))  _Float16 v8h;
typedef __attribute__((ext_vector_type(8)))  float    v8f;

#define B_      2
#define C_      128
#define H_      16
#define W_      512
#define HW_     (H_*W_)      /* 8192 */
#define D_      512
#define GROUPS_ 32
#define CG_     (C_/GROUPS_) /* 4 */
#define EPS_    1e-6f
#define SCALE_  0.08838834764831845f /* 1/sqrt(128) */

/* padded LDS strides (halves) to avoid 64-bank conflicts, 16B aligned */
#define KLD_ 136   /* 32 key rows x 128 ch, 272B stride */
#define VLD_ 40    /* 128 ch rows x 32 keys, 80B stride */
#define PLD_ 40    /* 16 q rows x 32 keys, 80B stride  */

/* ---- workspace layout (bytes) ---- */
#define OFF_TENC 0ull
#define OFF_W16  4096ull                         /* 4 x 128x128 f16 */
#define SZ_MAT   ((unsigned long long)B_*HW_*C_*2ull) /* 4 MB */
#define OFF_ET   (OFF_W16 + 4ull*16384ull*2ull)
#define OFF_QT   (OFF_ET + SZ_MAT)
#define OFF_KT   (OFF_QT + SZ_MAT)
#define OFF_V    (OFF_KT + SZ_MAT)
#define OFF_AT   (OFF_V  + SZ_MAT)

/* ================= CDNA5 async copy helpers ================= */
/* ASYNCcnt-tracked DMA of 16B global -> LDS (ISA 15.18.3 op 98).
   VDST operand is a VGPR holding the LDS byte address; low 32 bits of a
   generic LDS pointer are exactly that offset (aperture rules, ISA 10.2). */
__device__ __forceinline__ void async_ld_b128(void* lds, const void* g) {
  unsigned l = (unsigned)(unsigned long long)lds;
  unsigned long long a = (unsigned long long)g;
  asm volatile("global_load_async_to_lds_b128 %0, %1, off"
               :: "v"(l), "v"(a) : "memory");
}
__device__ __forceinline__ void wait_async0() {
  asm volatile("s_wait_asynccnt 0x0" ::: "memory");
}

/* ================= WMMA helpers ================= */
__device__ __forceinline__ v8f wmma16(v16h a, v16h b, v8f c) {
  return __builtin_amdgcn_wmma_f32_16x16x32_f16(
      /*neg_a=*/false, a, /*neg_b=*/false, b,
      /*c_mod=*/(short)0, c, /*reuse_a=*/false, /*reuse_b=*/false);
}

/* A fragment: 16x32 f16, row-major source, row stride ld (elements). */
__device__ __forceinline__ v16h load_a_frag(const _Float16* base, int ld, int lane) {
  int row = lane & 15, hf = lane >> 4;
  const _Float16* p = base + (size_t)row * ld + hf * 8;
  v8h lo = *(const v8h*)(p);
  v8h hi = *(const v8h*)(p + 16);
  v16h a;
#pragma unroll
  for (int i = 0; i < 8; ++i) { a[i] = lo[i]; a[i + 8] = hi[i]; }
  return a;
}

/* B fragment: 32x16 f16 column-major; memory holds B^T row-major, stride ld. */
__device__ __forceinline__ v16h load_b_frag(const _Float16* bt, long ld, int lane) {
  int col = lane & 15, koff = (lane >> 4) * 8;
  const _Float16* p = bt + (size_t)col * ld + koff;
  v8h lo = *(const v8h*)(p);
  v8h hi = *(const v8h*)(p + 16);
  v16h b;
#pragma unroll
  for (int i = 0; i < 8; ++i) { b[i] = lo[i]; b[i + 8] = hi[i]; }
  return b;
}

/* ================= 1. time embedding ================= */
__global__ void k_tenc(const int* __restrict__ t, const float* __restrict__ w,
                       const float* __restrict__ bias, float* __restrict__ tenc) {
  __shared__ float emb[D_];
  int b = blockIdx.x;
  int j = threadIdx.x; /* 0..511 */
  float tv = (float)t[b];
  int i = j >> 1;
  float arg = tv * __expf(-(2.0f * (float)i / (float)D_) * logf(10000.0f));
  float e = (j & 1) ? __cosf(arg) : __sinf(arg);
  emb[j] = fmaxf(e, 0.0f);
  __syncthreads();
  float acc = bias[j];
  const float* wr = w + (size_t)j * D_;
  for (int kk = 0; kk < D_; ++kk) acc = fmaf(emb[kk], wr[kk], acc);
  tenc[b * D_ + j] = acc;
}

/* ================= 2. weights -> f16 ================= */
__global__ void k_cvtw(const float* __restrict__ qw, const float* __restrict__ kw,
                       const float* __restrict__ vw, const float* __restrict__ pw,
                       _Float16* __restrict__ w16) {
  int idx = blockIdx.x * blockDim.x + threadIdx.x; /* < 65536 */
  int m = idx >> 14, r = idx & 16383;
  const float* s = (m == 0) ? qw : (m == 1) ? kw : (m == 2) ? vw : pw;
  w16[idx] = (_Float16)s[r];
}

/* ================= 3. broadcast add + GroupNorm -> eT (HW,C) f16 ======= */
__global__ void k_gn(const float* __restrict__ x, const float* __restrict__ tenc,
                     const float* __restrict__ gam, const float* __restrict__ bet,
                     _Float16* __restrict__ eT) {
  int b = blockIdx.x / GROUPS_, grp = blockIdx.x % GROUPS_;
  const int N = CG_ * H_ * W_; /* 32768 */
  __shared__ float s_sum[256], s_sq[256];
  float sum = 0.f, sq = 0.f;
  for (int idx = threadIdx.x; idx < N; idx += 256) {
    int cl = idx / (H_ * W_); int rem = idx % (H_ * W_); int wq = rem % W_;
    int c = grp * CG_ + cl;
    float v = x[((size_t)(b * C_ + c)) * HW_ + rem] + tenc[b * D_ + wq];
    sum += v; sq += v * v;
  }
  s_sum[threadIdx.x] = sum; s_sq[threadIdx.x] = sq;
  __syncthreads();
  for (int s = 128; s > 0; s >>= 1) {
    if (threadIdx.x < (unsigned)s) {
      s_sum[threadIdx.x] += s_sum[threadIdx.x + s];
      s_sq[threadIdx.x]  += s_sq[threadIdx.x + s];
    }
    __syncthreads();
  }
  float mu  = s_sum[0] / (float)N;
  float var = s_sq[0] / (float)N - mu * mu;
  float inv = rsqrtf(var + EPS_);
  for (int idx = threadIdx.x; idx < N; idx += 256) {
    int cl = idx / (H_ * W_); int rem = idx % (H_ * W_); int wq = rem % W_;
    int c = grp * CG_ + cl;
    float v = x[((size_t)(b * C_ + c)) * HW_ + rem] + tenc[b * D_ + wq];
    float y = (v - mu) * inv * gam[c] + bet[c];
    eT[(size_t)b * HW_ * C_ + (size_t)rem * C_ + c] = (_Float16)y;
  }
}

/* ================= 4. QKV 1x1 convs (WMMA) ================= */
__global__ __launch_bounds__(128) void
k_qkv(const _Float16* __restrict__ w16,
      const float* __restrict__ qb, const float* __restrict__ kb, const float* __restrict__ vb,
      const _Float16* __restrict__ eT,
      _Float16* __restrict__ qT, _Float16* __restrict__ kT, _Float16* __restrict__ vM) {
  int wid  = (int)((blockIdx.x * blockDim.x + threadIdx.x) >> 5); /* [0, 8192) */
  int lane = threadIdx.x & 31;
  int jt = wid & 511, ct = (wid >> 9) & 7, b = wid >> 12;
  int j0 = jt * 16, c0 = ct * 16;
  const _Float16* qw16 = w16;
  const _Float16* kw16 = w16 + 16384;
  const _Float16* vw16 = w16 + 32768;
  const _Float16* eb = eT + (size_t)b * HW_ * C_;
  v8f qa = {}, ka = {}, va = {};
#pragma unroll
  for (int cc = 0; cc < C_; cc += 32) {
    v16h bf = load_b_frag(eb + (size_t)j0 * C_ + cc, C_, lane);
    qa = wmma16(load_a_frag(qw16 + c0 * C_ + cc, C_, lane), bf, qa);
    ka = wmma16(load_a_frag(kw16 + c0 * C_ + cc, C_, lane), bf, ka);
    va = wmma16(load_a_frag(vw16 + c0 * C_ + cc, C_, lane), bf, va);
  }
  int n = lane & 15, hf = lane >> 4;
  v8h qs, ks;
#pragma unroll
  for (int r = 0; r < 8; ++r) {
    int c = c0 + r + 8 * hf;
    qs[r] = (_Float16)(qa[r] + qb[c]);
    ks[r] = (_Float16)(ka[r] + kb[c]);
  }
  size_t rowbase = (size_t)b * HW_ * C_ + (size_t)(j0 + n) * C_ + c0 + 8 * hf;
  *(v8h*)(qT + rowbase) = qs;    /* qT,kT stored (HW, C): A-frag ready */
  *(v8h*)(kT + rowbase) = ks;
#pragma unroll
  for (int r = 0; r < 8; ++r) {  /* v stored (C, HW): B-frag ready for PV */
    int c = c0 + r + 8 * hf;
    vM[(size_t)b * C_ * HW_ + (size_t)c * HW_ + j0 + n] = (_Float16)(va[r] + vb[c]);
  }
}

/* ================= 5. flash attention (WMMA + async-LDS staging) ======
   block = 4 waves = 64 queries sharing one 32-key K/V tile in LDS.
   K/V tiles double-buffered via global_load_async_to_lds_b128 (ASYNCcnt),
   so tile j+32 streams in while tile j is consumed from LDS.           */
__global__ __launch_bounds__(128) void
k_attn(const _Float16* __restrict__ qT, const _Float16* __restrict__ kT,
       const _Float16* __restrict__ vM, _Float16* __restrict__ attnT) {
  __shared__ __align__(16) _Float16 kbuf[2][32 * KLD_];   /* 17.0 KB */
  __shared__ __align__(16) _Float16 vbuf[2][128 * VLD_];  /* 20.0 KB */
  __shared__ __align__(16) _Float16 pbuf[4][16 * PLD_];   /*  5.0 KB */

  int tid = threadIdx.x;
  int wv = tid >> 5, lane = tid & 31;
  int b = blockIdx.x >> 7;                   /* 256 blocks: 128 per batch */
  int i0 = ((blockIdx.x & 127) << 6) + (wv << 4);
  const _Float16* qp = qT + (size_t)b * HW_ * C_;
  const _Float16* kp = kT + (size_t)b * HW_ * C_;
  const _Float16* vp = vM + (size_t)b * C_ * HW_;
  int n = lane & 15, hf = lane >> 4;

  /* resident Q A-fragments (16 queries x 128 ch) */
  v16h qa[4];
#pragma unroll
  for (int cc = 0; cc < 4; ++cc)
    qa[cc] = load_a_frag(qp + (size_t)i0 * C_ + cc * 32, C_, lane);

  float mrow[8], lrow[8];
  v8f O[8];
#pragma unroll
  for (int r = 0; r < 8; ++r) { mrow[r] = -INFINITY; lrow[r] = 0.f; }
#pragma unroll
  for (int t = 0; t < 8; ++t) { v8f z = {}; O[t] = z; }

  _Float16* myp = &pbuf[wv][0];

  /* cooperative async stage of one 32-key K tile + V tile (8KB + 8KB) */
  auto stage = [&](int p, int j) {
#pragma unroll
    for (int s = 0; s < 4; ++s) {            /* K: 512 x 16B chunks */
      int idx = tid + 128 * s;
      int row = idx >> 4, part = idx & 15;
      async_ld_b128(&kbuf[p][row * KLD_ + part * 8],
                    kp + (size_t)(j + row) * C_ + part * 8);
    }
#pragma unroll
    for (int s = 0; s < 4; ++s) {            /* V: 512 x 16B chunks */
      int idx = tid + 128 * s;
      int c = idx >> 2, part = idx & 3;
      async_ld_b128(&vbuf[p][c * VLD_ + part * 8],
                    vp + (size_t)c * HW_ + j + part * 8);
    }
  };

  int p = 0;
  stage(0, 0);
  wait_async0();
  __syncthreads();

  for (int j = 0; j < HW_; j += 32) {
    if (j + 32 < HW_) stage(p ^ 1, j + 32);  /* overlap next tile DMA */

    /* S tiles: 16 queries x 32 keys, K-dim = 128 ch, B-frags from LDS */
    v8f s0 = {}, s1 = {};
#pragma unroll
    for (int cc = 0; cc < 4; ++cc)
      s0 = wmma16(qa[cc], load_b_frag(&kbuf[p][cc * 32], KLD_, lane), s0);
#pragma unroll
    for (int cc = 0; cc < 4; ++cc)
      s1 = wmma16(qa[cc], load_b_frag(&kbuf[p][16 * KLD_ + cc * 32], KLD_, lane), s1);

    /* online softmax: row r of the D tile lives in VGPR r across a 16-lane
       group; after xor-reductions each lane holds stats for exactly the
       query rows its O accumulator owns. */
#pragma unroll
    for (int r = 0; r < 8; ++r) {
      float a0 = s0[r] * SCALE_, a1 = s1[r] * SCALE_;
      float mx = fmaxf(a0, a1);
      mx = fmaxf(mx, __shfl_xor(mx, 1, 32));
      mx = fmaxf(mx, __shfl_xor(mx, 2, 32));
      mx = fmaxf(mx, __shfl_xor(mx, 4, 32));
      mx = fmaxf(mx, __shfl_xor(mx, 8, 32));
      float newm = fmaxf(mrow[r], mx);
      float corr = __expf(mrow[r] - newm);
      mrow[r] = newm;
      float p0 = __expf(a0 - newm), p1 = __expf(a1 - newm);
      float rs = p0 + p1;
      rs += __shfl_xor(rs, 1, 32);
      rs += __shfl_xor(rs, 2, 32);
      rs += __shfl_xor(rs, 4, 32);
      rs += __shfl_xor(rs, 8, 32);
      lrow[r] = lrow[r] * corr + rs;
#pragma unroll
      for (int t = 0; t < 8; ++t) O[t][r] *= corr;
      int prow = r + 8 * hf;                 /* D-layout -> padded LDS rows */
      myp[prow * PLD_ + n]      = (_Float16)p0;
      myp[prow * PLD_ + 16 + n] = (_Float16)p1;
    }

    /* P (16x32) as A-fragment from per-wave LDS (DS in-order per wave) */
    v16h pa = load_a_frag(myp, PLD_, lane);

    /* O += P * V : 8 channel tiles, B-frags from LDS V tile */
#pragma unroll
    for (int t = 0; t < 8; ++t)
      O[t] = wmma16(pa, load_b_frag(&vbuf[p][(t * 16) * VLD_], VLD_, lane), O[t]);

    wait_async0();        /* next tile landed (own ops) ...      */
    __syncthreads();      /* ... and everyone done with tile p   */
    p ^= 1;
  }

  /* normalize + write attn output transposed (HW, C) f16 */
  _Float16* ob = attnT + (size_t)b * HW_ * C_;
#pragma unroll
  for (int r = 0; r < 8; ++r) {
    float inv = 1.0f / lrow[r];
    int row = i0 + r + 8 * hf;
#pragma unroll
    for (int t = 0; t < 8; ++t)
      ob[(size_t)row * C_ + t * 16 + n] = (_Float16)(O[t][r] * inv);
  }
}

/* ================= 6. output projection + residual (WMMA) ========= */
__global__ __launch_bounds__(128) void
k_proj(const _Float16* __restrict__ w16, const float* __restrict__ pb,
       const _Float16* __restrict__ attnT, const float* __restrict__ x,
       float* __restrict__ out) {
  int wid  = (int)((blockIdx.x * blockDim.x + threadIdx.x) >> 5);
  int lane = threadIdx.x & 31;
  int jt = wid & 511, ct = (wid >> 9) & 7, b = wid >> 12;
  int j0 = jt * 16, c0 = ct * 16;
  const _Float16* pw16 = w16 + 3 * 16384;
  const _Float16* ab = attnT + (size_t)b * HW_ * C_;
  v8f acc = {};
#pragma unroll
  for (int cc = 0; cc < C_; cc += 32) {
    v16h bf = load_b_frag(ab + (size_t)j0 * C_ + cc, C_, lane);
    acc = wmma16(load_a_frag(pw16 + c0 * C_ + cc, C_, lane), bf, acc);
  }
  int n = lane & 15, hf = lane >> 4;
#pragma unroll
  for (int r = 0; r < 8; ++r) {
    int c = c0 + r + 8 * hf;
    size_t idx = ((size_t)(b * C_ + c)) * HW_ + j0 + n;
    out[idx] = x[idx] + acc[r] + pb[c];
  }
}

/* ================= launch ================= */
extern "C" void kernel_launch(void* const* d_in, const int* in_sizes, int n_in,
                              void* d_out, int out_size, void* d_ws, size_t ws_size,
                              hipStream_t stream) {
  const float* x        = (const float*)d_in[0];
  const int*   t        = (const int*)  d_in[1];
  const float* t_proj_w = (const float*)d_in[2];
  const float* t_proj_b = (const float*)d_in[3];
  const float* gn_g     = (const float*)d_in[4];
  const float* gn_b     = (const float*)d_in[5];
  const float* qw = (const float*)d_in[6];  const float* qb = (const float*)d_in[7];
  const float* kw = (const float*)d_in[8];  const float* kb = (const float*)d_in[9];
  const float* vw = (const float*)d_in[10]; const float* vb = (const float*)d_in[11];
  const float* pw = (const float*)d_in[12]; const float* pb = (const float*)d_in[13];
  float* out = (float*)d_out;

  char* ws = (char*)d_ws;
  float*     tenc = (float*)(ws + OFF_TENC);
  _Float16*  w16  = (_Float16*)(ws + OFF_W16);
  _Float16*  eT   = (_Float16*)(ws + OFF_ET);
  _Float16*  qT   = (_Float16*)(ws + OFF_QT);
  _Float16*  kT   = (_Float16*)(ws + OFF_KT);
  _Float16*  vM   = (_Float16*)(ws + OFF_V);
  _Float16*  aT   = (_Float16*)(ws + OFF_AT);

  k_tenc<<<B_, D_, 0, stream>>>(t, t_proj_w, t_proj_b, tenc);
  k_cvtw<<<256, 256, 0, stream>>>(qw, kw, vw, pw, w16);
  k_gn<<<B_ * GROUPS_, 256, 0, stream>>>(x, tenc, gn_g, gn_b, eT);
  k_qkv<<<2048, 128, 0, stream>>>(w16, qb, kb, vb, eT, qT, kT, vM);
  k_attn<<<256, 128, 0, stream>>>(qT, kT, vM, aT);
  k_proj<<<2048, 128, 0, stream>>>(w16, pb, aT, x, out);
}
